// EfficientDGCNNBackbone_36481452213112
// MI455X (gfx1250) — compile-verified, matrix operand order checked by reading, not run on hardware
//
#include <hip/hip_runtime.h>

// ---------------------------------------------------------------------------
// EfficientDGCNN backbone for gfx1250 (MI455X).
// All matmuls (Gram/kNN, edge-conv 1x1, 512x512 dense, 1024->256 heads) run on
// V_WMMA_F32_16X16X32_BF16 (bf16 in, f32 accumulate). 8 waves/block; each wave
// register-blocks TWO 16x16 output tiles (one shared A fragment, two B/W
// fragments per k-chunk) to raise WMMA-per-load intensity.
// A/B fragments are hand-packed per the CDNA5 16-bit A-matrix VGPR layout
// (CDNA5 ISA §7.12.2):
//   lane L (<16):  row L,    elems 0-7 = K+0..7,  elems 8-15 = K+16..23
//   lane L (>=16): row L-16, elems 0-7 = K+8..15, elems 8-15 = K+24..31
// The same packing with "row" == output-column serves the B operand for
// row-major weight matrices W[O][Cin] (D = A x B, B[k][n] = W[n][k]).
// ---------------------------------------------------------------------------

typedef __bf16 bf16;
typedef __attribute__((ext_vector_type(16))) __bf16 v16bf;
typedef __attribute__((ext_vector_type(8)))  __bf16 v8bf;
typedef __attribute__((ext_vector_type(8)))  float  v8f;

#define K_NN   20
#define BATCH  2
#define NPTS   8192
#define BN_EPS 1e-5f
#define NEG_BIG (-3.0e38f)

__device__ inline bf16 f2bf(float f) {
  unsigned u = __builtin_bit_cast(unsigned, f);
  unsigned r = (u + 0x7FFFu + ((u >> 16) & 1u)) >> 16;   // round-to-nearest-even
  unsigned short s = (unsigned short)r;
  return __builtin_bit_cast(bf16, s);
}

// Load one 16x32 bf16 WMMA operand fragment from a row-major tile.
// base points at [row0][kc]; stride is the row pitch in elements.
__device__ inline v16bf load_frag(const bf16* base, int stride, int lane) {
  const bf16* p = base + (size_t)(lane & 15) * stride + ((lane >> 4) << 3);
  v8bf lo = *reinterpret_cast<const v8bf*>(p);        // K + {0..7} / {8..15}
  v8bf hi = *reinterpret_cast<const v8bf*>(p + 16);   // K + {16..23} / {24..31}
  v16bf f;
#pragma unroll
  for (int i = 0; i < 8; ++i) { f[i] = lo[i]; f[i + 8] = hi[i]; }
  return f;
}

__device__ inline v8f wmma_bf16(v16bf a, v16bf b, v8f c) {
  return __builtin_amdgcn_wmma_f32_16x16x32_bf16(false, a, false, b, (short)0, c,
                                                 false, false);
}

// Order-independent float atomic max (init with NEG_BIG).
__device__ inline void atomicMaxF(float* addr, float v) {
  if (v >= 0.f) atomicMax((int*)addr, __float_as_int(v));
  else          atomicMin((unsigned int*)addr, __float_as_uint(v));
}

// ---------------------------------------------------------------------------
// Small utility kernels
// ---------------------------------------------------------------------------

__global__ void transpose_x0_kernel(const float* __restrict__ x, float* __restrict__ x0t) {
  int t = blockIdx.x * 256 + threadIdx.x;                 // over B*N*4
  if (t >= BATCH * NPTS * 4) return;
  int c = t & 3, n = (t >> 2) & (NPTS - 1), b = t >> 15;
  x0t[t] = x[((size_t)b * 4 + c) * NPTS + n];
}

__global__ void cvt_pad_kernel(const float* __restrict__ src, int rowStride, int chOff,
                               long long rows, int cols, int colsPad,
                               bf16* __restrict__ dst) {
  long long t = (long long)blockIdx.x * 256 + threadIdx.x;
  if (t >= rows * colsPad) return;
  long long r = t / colsPad; int c = (int)(t % colsPad);
  float v = (c < cols) ? src[r * (long long)rowStride + chOff + c] : 0.f;
  dst[t] = f2bf(v);
}

__global__ void rownorm_kernel(const float* __restrict__ src, int rowStride, int chOff,
                               int C, int rows, float* __restrict__ xx) {
  int r = blockIdx.x * 256 + threadIdx.x;
  if (r >= rows) return;
  const float* p = src + (size_t)r * rowStride + chOff;
  float s = 0.f;
  for (int c = 0; c < C; ++c) { float v = p[c]; s += v * v; }
  xx[r] = s;
}

__global__ void fill_f32_kernel(float* __restrict__ p, long long n, float v) {
  long long t = (long long)blockIdx.x * 256 + threadIdx.x;
  if (t < n) p[t] = v;
}

__global__ void fill_channels_kernel(float* __restrict__ p, int rowStride, int chOff,
                                     int nch, long long rows, float v) {
  long long t = (long long)blockIdx.x * 256 + threadIdx.x;
  if (t >= rows * nch) return;
  long long r = t / nch; int c = (int)(t % nch);
  p[r * rowStride + chOff + c] = v;
}

// ---------------------------------------------------------------------------
// Fused kNN: per block 128 query rows, per wave 16 rows x 32 columns per step
// (two WMMA tiles sharing one A fragment). neg_d = 2*dot - |xi|^2 - |xj|^2,
// top-K=20 kept as sorted insertion lists in LDS.
// ---------------------------------------------------------------------------
__global__ void knn_kernel(const bf16* __restrict__ xbf, int Cp,
                           const float* __restrict__ xx, int* __restrict__ idx) {
  extern __shared__ char smem[];
  float* topv  = (float*)smem;                          // [128][K_NN]
  int*   topi  = (int*)(smem + 128 * K_NN * 4);         // [128][K_NN]
  float* stage = (float*)(smem + 2 * 128 * K_NN * 4);   // [128][33]

  const int tid = threadIdx.x, lane = tid & 31, wave = tid >> 5;
  const int b = blockIdx.y;
  const int row0 = blockIdx.x * 128;
  const bf16*  xb  = xbf + (size_t)b * NPTS * Cp;
  const float* xxb = xx  + (size_t)b * NPTS;

  for (int t = tid; t < 128 * K_NN; t += 256) { topv[t] = NEG_BIG; topi[t] = 0; }
  __syncthreads();

  const bf16* abase = xb + (size_t)(row0 + wave * 16) * Cp;
  for (int j0 = 0; j0 < NPTS; j0 += 32) {
    if (lane == 0 && j0 + 32 < NPTS)
      __builtin_prefetch(xb + (size_t)(j0 + 32) * Cp, 0, 1);  // global_prefetch_b8
    const bf16* bbase0 = xb + (size_t)j0 * Cp;
    const bf16* bbase1 = xb + (size_t)(j0 + 16) * Cp;
    v8f acc0 = {}, acc1 = {};
    for (int kc = 0; kc < Cp; kc += 32) {
      v16bf a = load_frag(abase + kc, Cp, lane);          // shared A fragment
      acc0 = wmma_bf16(a, load_frag(bbase0 + kc, Cp, lane), acc0);
      acc1 = wmma_bf16(a, load_frag(bbase1 + kc, Cp, lane), acc1);
    }
    const int colL  = lane & 15;
    const int rhalf = (lane >> 4) << 3;
    const float xxc0 = xxb[j0 + colL];
    const float xxc1 = xxb[j0 + 16 + colL];
#pragma unroll
    for (int v = 0; v < 8; ++v) {
      int rloc = v + rhalf;
      float xxr = xxb[row0 + wave * 16 + rloc];
      stage[(wave * 16 + rloc) * 33 + colL]      = 2.f * acc0[v] - xxr - xxc0;
      stage[(wave * 16 + rloc) * 33 + 16 + colL] = 2.f * acc1[v] - xxr - xxc1;
    }
    __syncthreads();
    if (lane < 16) {
      int r = wave * 16 + lane;
      float* tv = topv + r * K_NN;
      int*   ti = topi + r * K_NN;
      for (int cc = 0; cc < 32; ++cc) {
        float v = stage[r * 33 + cc];
        if (v > tv[0]) {
          int p = 0;
          while (p < K_NN - 1 && tv[p + 1] < v) { tv[p] = tv[p + 1]; ti[p] = ti[p + 1]; ++p; }
          tv[p] = v; ti[p] = j0 + cc;
        }
      }
    }
    __syncthreads();
  }
  for (int t = tid; t < 128 * K_NN; t += 256) {
    int r = t / K_NN, k = t - r * K_NN;
    idx[((size_t)b * NPTS + row0 + r) * K_NN + k] = topi[r * K_NN + (K_NN - 1 - k)];
  }
}

// ---------------------------------------------------------------------------
// Edge conv: M = N*K rows of feat = [x_nb - x_n, x_n] (gathered into LDS bf16),
// h = feat @ W^T via WMMA, two output tiles per wave per step.
// pass 0: per-channel sum/sumsq (BN stats).
// pass 1: y = a*h + c, leaky(0.2), max over K via float atomic max.
// ---------------------------------------------------------------------------
__global__ void edgeconv_kernel(const float* __restrict__ X, int rowStride, int chOff,
                                int C, int C2p, const int* __restrict__ idx,
                                const bf16* __restrict__ Wbf, int O,
                                float* __restrict__ sums, float* __restrict__ sumsq,
                                const float* __restrict__ aCoef, const float* __restrict__ cCoef,
                                float* __restrict__ out, int outStride, int outChOff,
                                int pass) {
  extern __shared__ char smem[];
  bf16*  feat = (bf16*)smem;                              // [128][C2p]
  float* lsum = (float*)(smem + (size_t)128 * C2p * 2);   // [O]
  float* lsq  = lsum + O;                                 // [O]

  const int tid = threadIdx.x, lane = tid & 31, wave = tid >> 5;
  const int b  = blockIdx.y;
  const int m0 = blockIdx.x * 128;
  const int C2 = 2 * C;
  const float* Xb   = X + (size_t)b * NPTS * rowStride + chOff;
  const int*   idxb = idx + (size_t)b * NPTS * K_NN;

  if (pass == 0) for (int i = tid; i < 2 * O; i += 256) lsum[i] = 0.f;

  for (int t = tid; t < 128 * C2p; t += 256) {
    int r = t / C2p, cc = t - r * C2p;
    int m = m0 + r;
    int n = m / K_NN, kk = m - n * K_NN;
    float v = 0.f;
    if (cc < C2) {
      if (cc < C) {
        int nb = idxb[(size_t)n * K_NN + kk];
        v = Xb[(size_t)nb * rowStride + cc] - Xb[(size_t)n * rowStride + cc];
      } else {
        v = Xb[(size_t)n * rowStride + (cc - C)];
      }
    }
    feat[t] = f2bf(v);
  }
  __syncthreads();

  const bf16* ab = feat + (size_t)(wave * 16) * C2p;
  for (int ot = 0; ot < O; ot += 32) {                    // O is a multiple of 32
    v8f acc0 = {}, acc1 = {};
    const bf16* wb0 = Wbf + (size_t)ot * C2p;
    const bf16* wb1 = Wbf + (size_t)(ot + 16) * C2p;
    for (int kc = 0; kc < C2p; kc += 32) {
      v16bf a = load_frag(ab + kc, C2p, lane);            // shared A fragment
      acc0 = wmma_bf16(a, load_frag(wb0 + kc, C2p, lane), acc0);
      acc1 = wmma_bf16(a, load_frag(wb1 + kc, C2p, lane), acc1);
    }
    const int col0  = ot + (lane & 15);
    const int col1  = col0 + 16;
    const int rhalf = (lane >> 4) << 3;
    if (pass == 0) {
      float s0 = 0.f, q0 = 0.f, s1 = 0.f, q1 = 0.f;
#pragma unroll
      for (int v = 0; v < 8; ++v) {
        s0 += acc0[v]; q0 += acc0[v] * acc0[v];
        s1 += acc1[v]; q1 += acc1[v] * acc1[v];
      }
      atomicAdd(&lsum[col0], s0); atomicAdd(&lsq[col0], q0);
      atomicAdd(&lsum[col1], s1); atomicAdd(&lsq[col1], q1);
    } else {
      float a0 = aCoef[col0], c0 = cCoef[col0];
      float a1 = aCoef[col1], c1 = cCoef[col1];
#pragma unroll
      for (int v = 0; v < 8; ++v) {
        int m = m0 + wave * 16 + v + rhalf;
        int n = m / K_NN;
        float* row = &out[((size_t)b * NPTS + n) * outStride + outChOff];
        float y0 = a0 * acc0[v] + c0; y0 = fmaxf(y0, 0.2f * y0);   // leaky 0.2
        float y1 = a1 * acc1[v] + c1; y1 = fmaxf(y1, 0.2f * y1);
        atomicMaxF(row + col0, y0);
        atomicMaxF(row + col1, y1);
      }
    }
  }
  if (pass == 0) {
    __syncthreads();
    for (int i = tid; i < O; i += 256) {
      atomicAdd(&sums[i],  lsum[i]);
      atomicAdd(&sumsq[i], lsq[i]);
    }
  }
}

// ---------------------------------------------------------------------------
// Dense GEMM (A[M][Kp] bf16, W[O][Kp] bf16), streamed fragments from global,
// two output tiles per wave per step.
// pass 0: BN stats per group (group = row / rowsPerGroup). pass 1: normalize
// (+ optional leaky 0.2) and store to out with row stride / channel offset.
// Bias of the head layers is omitted: per-sample BN subtracts the mean, so a
// constant channel bias cancels exactly before gamma/beta.
// ---------------------------------------------------------------------------
__global__ void gemm_kernel(const bf16* __restrict__ Abf, int Kp,
                            const bf16* __restrict__ Wbf, int O,
                            float* __restrict__ sums, float* __restrict__ sumsq,
                            const float* __restrict__ aCoef, const float* __restrict__ cCoef,
                            float* __restrict__ out, int outStride, int outChOff,
                            int leaky, int rowsPerGroup, int pass) {
  extern __shared__ char smem[];
  float* lsum = (float*)smem;     // [O]
  float* lsq  = lsum + O;         // [O]

  const int tid = threadIdx.x, lane = tid & 31, wave = tid >> 5;
  const int m0 = blockIdx.x * 128;
  const int g  = m0 / rowsPerGroup;

  if (pass == 0) {
    for (int i = tid; i < 2 * O; i += 256) lsum[i] = 0.f;
    __syncthreads();
  }

  const bf16* ab0 = Abf + (size_t)(m0 + wave * 16) * Kp;
  for (int ot = 0; ot < O; ot += 32) {                    // O is a multiple of 32
    v8f acc0 = {}, acc1 = {};
    const bf16* wb0 = Wbf + (size_t)ot * Kp;
    const bf16* wb1 = Wbf + (size_t)(ot + 16) * Kp;
    for (int kc = 0; kc < Kp; kc += 32) {
      v16bf a = load_frag(ab0 + kc, Kp, lane);            // shared A fragment
      acc0 = wmma_bf16(a, load_frag(wb0 + kc, Kp, lane), acc0);
      acc1 = wmma_bf16(a, load_frag(wb1 + kc, Kp, lane), acc1);
    }
    const int col0  = ot + (lane & 15);
    const int col1  = col0 + 16;
    const int rhalf = (lane >> 4) << 3;
    if (pass == 0) {
      float s0 = 0.f, q0 = 0.f, s1 = 0.f, q1 = 0.f;
#pragma unroll
      for (int v = 0; v < 8; ++v) {
        s0 += acc0[v]; q0 += acc0[v] * acc0[v];
        s1 += acc1[v]; q1 += acc1[v] * acc1[v];
      }
      atomicAdd(&lsum[col0], s0); atomicAdd(&lsq[col0], q0);
      atomicAdd(&lsum[col1], s1); atomicAdd(&lsq[col1], q1);
    } else {
      float a0 = aCoef[g * O + col0], c0 = cCoef[g * O + col0];
      float a1 = aCoef[g * O + col1], c1 = cCoef[g * O + col1];
#pragma unroll
      for (int v = 0; v < 8; ++v) {
        int m = m0 + wave * 16 + v + rhalf;
        float* row = &out[(size_t)m * outStride + outChOff];
        float y0 = a0 * acc0[v] + c0;
        float y1 = a1 * acc1[v] + c1;
        if (leaky) { y0 = fmaxf(y0, 0.2f * y0); y1 = fmaxf(y1, 0.2f * y1); }
        row[col0] = y0;
        row[col1] = y1;
      }
    }
  }
  if (pass == 0) {
    __syncthreads();
    for (int i = tid; i < O; i += 256) {
      atomicAdd(&sums[g * O + i],  lsum[i]);
      atomicAdd(&sumsq[g * O + i], lsq[i]);
    }
  }
}

__global__ void bn_finalize_kernel(const float* __restrict__ sums, const float* __restrict__ sumsq,
                                   const float* __restrict__ gamma, const float* __restrict__ beta,
                                   float invCount, int O, int total,
                                   float* __restrict__ aCoef, float* __restrict__ cCoef) {
  int i = blockIdx.x * 256 + threadIdx.x;
  if (i >= total) return;
  int o = i % O;
  float m   = sums[i] * invCount;
  float var = sumsq[i] * invCount - m * m;
  float inv = rsqrtf(var + BN_EPS);
  float gm  = gamma[o];
  aCoef[i] = gm * inv;
  cCoef[i] = beta[o] - m * gm * inv;
}

__global__ void gmax_kernel(const float* __restrict__ h, float* __restrict__ gmax) {
  extern __shared__ char smem[];
  float* red = (float*)smem;
  int o = blockIdx.x, b = blockIdx.y;
  const float* p = h + (size_t)b * NPTS * 1024 + o;
  float m = NEG_BIG;
  for (int n = threadIdx.x; n < NPTS; n += 128) m = fmaxf(m, p[(size_t)n * 1024]);
  red[threadIdx.x] = m;
  __syncthreads();
  for (int s = 64; s > 0; s >>= 1) {
    if (threadIdx.x < s) red[threadIdx.x] = fmaxf(red[threadIdx.x], red[threadIdx.x + s]);
    __syncthreads();
  }
  if (threadIdx.x == 0) gmax[b * 512 + o] = red[0];
}

__global__ void bcast_kernel(const float* __restrict__ gmax, float* __restrict__ h) {
  long long t = (long long)blockIdx.x * 256 + threadIdx.x;   // B*N*512
  if (t >= (long long)BATCH * NPTS * 512) return;
  int o = (int)(t % 512); long long bn = t / 512; int b = (int)(bn / NPTS);
  h[bn * 1024 + 512 + o] = gmax[b * 512 + o];
}

__global__ void logits_kernel(const float* __restrict__ Ws, const float* __restrict__ bs,
                              float* __restrict__ out) {
  long long t = (long long)blockIdx.x * 256 + threadIdx.x;   // B*N*20
  if (t >= (long long)BATCH * NPTS * 20) return;
  int o = (int)(t % 20); long long bn = t / 20;
  const float* f = out + bn * 1044 + 768;                    // levels[3]
  const float* w = Ws + o * 256;
  float s = bs[o];
  for (int c = 0; c < 256; ++c) s += f[c] * w[c];
  out[bn * 1044 + 1024 + o] = s;
}

// ---------------------------------------------------------------------------
// Host orchestration
// ---------------------------------------------------------------------------
extern "C" void kernel_launch(void* const* d_in, const int* in_sizes, int n_in,
                              void* d_out, int out_size, void* d_ws, size_t ws_size,
                              hipStream_t stream) {
  (void)in_sizes; (void)n_in; (void)out_size; (void)ws_size;

  const float* x   = (const float*)d_in[0];
  const float* Win[4]  = {(const float*)d_in[1], (const float*)d_in[4],
                          (const float*)d_in[7], (const float*)d_in[10]};
  const float* gin[4]  = {(const float*)d_in[2], (const float*)d_in[5],
                          (const float*)d_in[8], (const float*)d_in[11]};
  const float* bin[4]  = {(const float*)d_in[3], (const float*)d_in[6],
                          (const float*)d_in[9], (const float*)d_in[12]};
  const float* W5 = (const float*)d_in[13];
  const float* g5 = (const float*)d_in[14];
  const float* b5 = (const float*)d_in[15];
  const float* Wf = (const float*)d_in[16];
  const float* gf = (const float*)d_in[18];
  const float* betaf = (const float*)d_in[19];
  const float* Ws = (const float*)d_in[20];
  const float* bs = (const float*)d_in[21];
  float* out = (float*)d_out;
  char* ws = (char*)d_ws;

  // ---- workspace layout (~145 MB) ----
  size_t off = 0;
  auto alloc = [&](size_t bytes) { size_t o = off; off = (off + bytes + 255) & ~(size_t)255; return o; };
  const long long rowsBN = (long long)BATCH * NPTS;
  int*   s_idx  = (int*)  (ws + alloc((size_t)BATCH * NPTS * K_NN * 4));
  float* s_xx   = (float*)(ws + alloc((size_t)BATCH * NPTS * 4));
  bf16*  s_xbf  = (bf16*) (ws + alloc((size_t)BATCH * NPTS * 128 * 2));
  float* s_xc   = (float*)(ws + alloc((size_t)BATCH * NPTS * 512 * 4));   // concat x1..x4
  float* s_h    = (float*)(ws + alloc((size_t)BATCH * NPTS * 1024 * 4));  // h5 | gmax
  bf16*  s_hbf  = (bf16*) (ws + alloc((size_t)BATCH * NPTS * 1024 * 2));
  bf16*  s_wbf  = (bf16*) (ws + alloc((size_t)(64*32 + 64*128 + 128*128 + 256*256 + 512*512 + 1024*1024) * 2));
  float* s_sum  = (float*)(ws + alloc((size_t)BATCH * 1024 * 4));
  float* s_sq   = (float*)(ws + alloc((size_t)BATCH * 1024 * 4));
  float* s_a    = (float*)(ws + alloc((size_t)BATCH * 1024 * 4));
  float* s_c    = (float*)(ws + alloc((size_t)BATCH * 1024 * 4));
  float* s_gmax = (float*)(ws + alloc((size_t)BATCH * 512 * 4));
  float* s_x0t  = (float*)(ws + alloc((size_t)BATCH * NPTS * 4 * 4));

  bf16* w1bf = s_wbf;
  bf16* w2bf = w1bf + 64 * 32;
  bf16* w3bf = w2bf + 64 * 128;
  bf16* w4bf = w3bf + 128 * 128;
  bf16* w5bf = w4bf + 256 * 256;
  bf16* wfbf = w5bf + 512 * 512;

  auto blocks1d = [](long long n) { return (unsigned)((n + 255) / 256); };

  // ---- x -> [B,N,4], weights -> padded bf16 ----
  transpose_x0_kernel<<<blocks1d(rowsBN * 4), 256, 0, stream>>>(x, s_x0t);
  auto cvtW = [&](const float* src, int rows, int cols, int pad, bf16* dst) {
    cvt_pad_kernel<<<blocks1d((long long)rows * pad), 256, 0, stream>>>(src, cols, 0, rows, cols, pad, dst);
  };
  cvtW(Win[0], 64, 8, 32, w1bf);
  cvtW(Win[1], 64, 128, 128, w2bf);
  cvtW(Win[2], 128, 128, 128, w3bf);
  cvtW(Win[3], 256, 256, 256, w4bf);
  cvtW(W5, 512, 512, 512, w5bf);
  cvtW(Wf, 1024, 1024, 1024, wfbf);   // 4 stacked [256][1024] blocks

  // ---- edge conv layers ----
  const float* srcPtr[4] = { s_x0t, s_xc, s_xc, s_xc };
  const int srcStride[4] = { 4, 512, 512, 512 };
  const int srcOff[4]    = { 0, 0, 64, 128 };
  const int Cs[4]        = { 4, 64, 64, 128 };
  const int Cps[4]       = { 32, 64, 64, 128 };
  const int C2ps[4]      = { 32, 128, 128, 256 };
  const int Os[4]        = { 64, 64, 128, 256 };
  const int outOff[4]    = { 0, 64, 128, 256 };
  bf16* wbfs[4] = { w1bf, w2bf, w3bf, w4bf };

  for (int L = 0; L < 4; ++L) {
    const float* src = srcPtr[L];
    const int st = srcStride[L], so = srcOff[L];
    const int C = Cs[L], Cp = Cps[L], C2p = C2ps[L], O = Os[L];

    cvt_pad_kernel<<<blocks1d(rowsBN * Cp), 256, 0, stream>>>(src, st, so, rowsBN, C, Cp, s_xbf);
    rownorm_kernel<<<blocks1d(rowsBN), 256, 0, stream>>>(src, st, so, C, (int)rowsBN, s_xx);

    dim3 kg(NPTS / 128, BATCH);
    size_t kshm = (size_t)2 * 128 * K_NN * 4 + (size_t)128 * 33 * 4;
    knn_kernel<<<kg, 256, kshm, stream>>>(s_xbf, Cp, s_xx, s_idx);

    fill_f32_kernel<<<blocks1d(O), 256, 0, stream>>>(s_sum, O, 0.f);
    fill_f32_kernel<<<blocks1d(O), 256, 0, stream>>>(s_sq, O, 0.f);

    dim3 eg(NPTS * K_NN / 128, BATCH);
    size_t eshm = (size_t)128 * C2p * 2 + (size_t)2 * O * 4;
    edgeconv_kernel<<<eg, 256, eshm, stream>>>(src, st, so, C, C2p, s_idx, wbfs[L], O,
                                               s_sum, s_sq, nullptr, nullptr,
                                               nullptr, 0, 0, /*pass=*/0);
    bn_finalize_kernel<<<1, 256, 0, stream>>>(s_sum, s_sq, gin[L], bin[L],
                                              1.0f / ((float)BATCH * NPTS * K_NN), O, O, s_a, s_c);
    fill_channels_kernel<<<blocks1d(rowsBN * O), 256, 0, stream>>>(s_xc, 512, outOff[L], O, rowsBN, NEG_BIG);
    edgeconv_kernel<<<eg, 256, eshm, stream>>>(src, st, so, C, C2p, s_idx, wbfs[L], O,
                                               nullptr, nullptr, s_a, s_c,
                                               s_xc, 512, outOff[L], /*pass=*/1);
  }

  // ---- h5 = leaky(BN(xc @ W5^T)) into s_h[:, :512] ----
  cvt_pad_kernel<<<blocks1d(rowsBN * 512), 256, 0, stream>>>(s_xc, 512, 0, rowsBN, 512, 512, s_hbf);
  fill_f32_kernel<<<blocks1d(512), 256, 0, stream>>>(s_sum, 512, 0.f);
  fill_f32_kernel<<<blocks1d(512), 256, 0, stream>>>(s_sq, 512, 0.f);
  dim3 gg((unsigned)(rowsBN / 128));
  gemm_kernel<<<gg, 256, 2 * 512 * 4, stream>>>(s_hbf, 512, w5bf, 512, s_sum, s_sq,
                                                nullptr, nullptr, nullptr, 0, 0, 0,
                                                BATCH * NPTS, /*pass=*/0);
  bn_finalize_kernel<<<2, 256, 0, stream>>>(s_sum, s_sq, g5, b5,
                                            1.0f / ((float)BATCH * NPTS), 512, 512, s_a, s_c);
  gemm_kernel<<<gg, 256, 2 * 512 * 4, stream>>>(s_hbf, 512, w5bf, 512, nullptr, nullptr,
                                                s_a, s_c, s_h, 1024, 0, /*leaky=*/1,
                                                BATCH * NPTS, /*pass=*/1);

  // ---- global max over points, concat into s_h[:, 512:1024] ----
  dim3 gm(512, BATCH);
  gmax_kernel<<<gm, 128, 128 * 4, stream>>>(s_h, s_gmax);
  bcast_kernel<<<blocks1d(rowsBN * 512), 256, 0, stream>>>(s_gmax, s_h);
  cvt_pad_kernel<<<blocks1d(rowsBN * 1024), 256, 0, stream>>>(s_h, 1024, 0, rowsBN, 1024, 1024, s_hbf);

  // ---- 4 feature heads (per-sample BN), written straight into d_out ----
  for (int i = 0; i < 4; ++i) {
    fill_f32_kernel<<<blocks1d(BATCH * 256), 256, 0, stream>>>(s_sum, BATCH * 256, 0.f);
    fill_f32_kernel<<<blocks1d(BATCH * 256), 256, 0, stream>>>(s_sq, BATCH * 256, 0.f);
    const bf16* wi = wfbf + (size_t)i * 256 * 1024;
    gemm_kernel<<<gg, 256, 2 * 256 * 4, stream>>>(s_hbf, 1024, wi, 256, s_sum, s_sq,
                                                  nullptr, nullptr, nullptr, 0, 0, 0,
                                                  NPTS, /*pass=*/0);
    bn_finalize_kernel<<<blocks1d(BATCH * 256), 256, 0, stream>>>(
        s_sum, s_sq, gf + i * 256, betaf + i * 256, 1.0f / (float)NPTS, 256, BATCH * 256, s_a, s_c);
    gemm_kernel<<<gg, 256, 2 * 256 * 4, stream>>>(s_hbf, 1024, wi, 256, nullptr, nullptr,
                                                  s_a, s_c, out, 1044, i * 256, /*leaky=*/0,
                                                  NPTS, /*pass=*/1);
  }

  // ---- classifier logits into d_out[:, 1024:1044] ----
  logits_kernel<<<blocks1d(rowsBN * 20), 256, 0, stream>>>(Ws, bs, out);
}
